// GIC_33054068310374
// MI455X (gfx1250) — compile-verified, matrix operand order checked by reading, not run on hardware
//
#include <hip/hip_runtime.h>
#include <hip/hip_bf16.h>

typedef __attribute__((ext_vector_type(16))) __bf16 v16bf;
typedef __attribute__((ext_vector_type(8)))  __bf16 v8bf;
typedef __attribute__((ext_vector_type(8)))  float  v8f;

#define HID   512
#define DIMF  512
#define KCL   128
#define CITER 11
#define BETA_T 100.0f

static __device__ __forceinline__ float softplusf(float x) {
  return log1pf(__expf(-fabsf(x))) + fmaxf(x, 0.0f);
}

// ---------------------------------------------------------------------------
// WMMA fragment loader: A 16x32 bf16 fragment from a row-major source.
//   lane L: row = L&15, half = L>>4; elements 0..7 = K[k0+half*8 .. +7],
//   elements 8..15 = K[k0+16+half*8 .. +7]  -> two contiguous 16B loads.
// The B fragment uses the identical map when B is stored transposed
// (Bt[n][k] row-major), so one loader serves both.
// ---------------------------------------------------------------------------
__device__ __forceinline__ v16bf load_frag(const __bf16* __restrict__ base,
                                           int ld, int k0, int lane) {
  int r    = lane & 15;
  int half = lane >> 4;
  const __bf16* p = base + (size_t)r * ld + k0 + half * 8;
  union { v16bf v; v8bf h[2]; } u;
  u.h[0] = *(const v8bf*)(p);
  u.h[1] = *(const v8bf*)(p + 16);
  return u.v;
}

#define WMMA_BF16(A_, B_, C_) \
  __builtin_amdgcn_wmma_f32_16x16x32_bf16(false, (A_), false, (B_), (short)0, (C_), false, false)

// Epilogue modes: 0 store, 1 relu(x+bias[n]), 2 x+bias[n], 3 x*rowdiv[m]
__device__ __forceinline__ void store_tile(float* __restrict__ C, int ldc,
                                           int m0, int n, v8f c,
                                           const float* __restrict__ bias,
                                           const float* __restrict__ rowdiv,
                                           int mode) {
  float bv = bias ? bias[n] : 0.0f;
  #pragma unroll
  for (int i = 0; i < 8; ++i) {
    float v = c[i];
    int m = m0 + i;
    if      (mode == 1) { v += bv; v = fmaxf(v, 0.0f); }
    else if (mode == 2) { v += bv; }
    else if (mode == 3) { v *= rowdiv[m]; }
    C[(size_t)m * ldc + n] = v;
  }
}

// ---------------------------------------------------------------------------
// bf16 GEMM, 2x2 register-blocked + software-pipelined.
// C[M,Ncol](f32) = A[M,Kd](bf16,lda) @ Bt[Ncol,Kd](bf16,ldb)^T
// One wave per 32x32 C supertile: 4 C fragments, 4 WMMA per k-step on
// 2 A + 2 B fragments; next step's fragments are loaded before the WMMAs
// so vmem latency overlaps matrix math. M, Ncol, Kd multiples of 32.
// ---------------------------------------------------------------------------
__global__ void gemm_bf16_kernel(const __bf16* __restrict__ A, int lda,
                                 const __bf16* __restrict__ Bt, int ldb,
                                 float* __restrict__ C, int ldc,
                                 int M, int Ncol, int Kd,
                                 const float* __restrict__ bias,
                                 const float* __restrict__ rowdiv, int mode)
{
  int tilesN = Ncol >> 5;
  int total  = (M >> 5) * tilesN;
  int wave   = blockIdx.x * (blockDim.x >> 5) + (threadIdx.x >> 5);
  if (wave >= total) return;
  int tm = wave / tilesN;
  int tn = wave - tm * tilesN;
  const __bf16* A0 = A  + (size_t)(tm << 5) * lda;
  const __bf16* A1 = A0 + (size_t)16 * lda;
  const __bf16* B0 = Bt + (size_t)(tn << 5) * ldb;
  const __bf16* B1 = B0 + (size_t)16 * ldb;
  int lane = threadIdx.x & 31;
  int prow = lane & 15;

  v8f c00 = {}, c01 = {}, c10 = {}, c11 = {};
  v16bf a0 = load_frag(A0, lda, 0, lane);
  v16bf a1 = load_frag(A1, lda, 0, lane);
  v16bf b0 = load_frag(B0, ldb, 0, lane);
  v16bf b1 = load_frag(B1, ldb, 0, lane);

  int k0 = 0;
  for (; k0 + 32 < Kd; k0 += 32) {
    int kn = k0 + 32;
    if (kn + 32 < Kd) {  // prefetch two steps ahead -> global_prefetch_b8
      __builtin_prefetch(A0 + (size_t)prow * lda + kn + 32, 0, 3);
      __builtin_prefetch(A1 + (size_t)prow * lda + kn + 32, 0, 3);
      __builtin_prefetch(B0 + (size_t)prow * ldb + kn + 32, 0, 3);
      __builtin_prefetch(B1 + (size_t)prow * ldb + kn + 32, 0, 3);
    }
    // issue next step's loads before consuming current fragments
    v16bf an0 = load_frag(A0, lda, kn, lane);
    v16bf an1 = load_frag(A1, lda, kn, lane);
    v16bf bn0 = load_frag(B0, ldb, kn, lane);
    v16bf bn1 = load_frag(B1, ldb, kn, lane);
    c00 = WMMA_BF16(a0, b0, c00);
    c01 = WMMA_BF16(a0, b1, c01);
    c10 = WMMA_BF16(a1, b0, c10);
    c11 = WMMA_BF16(a1, b1, c11);
    a0 = an0; a1 = an1; b0 = bn0; b1 = bn1;
  }
  // drain
  c00 = WMMA_BF16(a0, b0, c00);
  c01 = WMMA_BF16(a0, b1, c01);
  c10 = WMMA_BF16(a1, b0, c10);
  c11 = WMMA_BF16(a1, b1, c11);

  int nl = lane & 15;
  int mb = (lane >> 4) << 3;
  int m0 = (tm << 5) + mb;
  int n0 = (tn << 5) + nl;
  store_tile(C, ldc, m0,      n0,      c00, bias, rowdiv, mode);
  store_tile(C, ldc, m0,      n0 + 16, c01, bias, rowdiv, mode);
  store_tile(C, ldc, m0 + 16, n0,      c10, bias, rowdiv, mode);
  store_tile(C, ldc, m0 + 16, n0 + 16, c11, bias, rowdiv, mode);
}

// ---------------------------------------------------------------------------
// Fused cluster discriminator: cs = sigmoid(r @ mu); accumulate
// sum softplus(-dot(pos_row,cs_row)) and sum softplus(dot(neg_row,cs_row)).
// One wave per 16-node row tile; loops over 32 column tiles of H=512.
// ---------------------------------------------------------------------------
__global__ void cluster_loss_kernel(const __bf16* __restrict__ r,
                                    const __bf16* __restrict__ muT,
                                    const float* __restrict__ pos,
                                    const float* __restrict__ neg,
                                    float* __restrict__ acc, int Nn)
{
  int tiles = Nn >> 4;
  int tile  = blockIdx.x * (blockDim.x >> 5) + (threadIdx.x >> 5);
  if (tile >= tiles) return;
  int row0   = tile << 4;
  int lane   = threadIdx.x & 31;
  int nloc   = lane & 15;
  int mbase  = (lane >> 4) << 3;
  const __bf16* Abase = r + (size_t)row0 * KCL;
  // A fragments are reused across all 32 column tiles: load once.
  v16bf a0 = load_frag(Abase, KCL, 0,  lane);
  v16bf a1 = load_frag(Abase, KCL, 32, lane);
  v16bf a2 = load_frag(Abase, KCL, 64, lane);
  v16bf a3 = load_frag(Abase, KCL, 96, lane);
  float accp[8], accn[8];
  #pragma unroll
  for (int i = 0; i < 8; ++i) { accp[i] = 0.f; accn[i] = 0.f; }

  for (int ct = 0; ct < (HID >> 4); ++ct) {
    const __bf16* Bbase = muT + (size_t)(ct << 4) * KCL;
    v8f c = {};
    c = WMMA_BF16(a0, load_frag(Bbase, KCL, 0,  lane), c);
    c = WMMA_BF16(a1, load_frag(Bbase, KCL, 32, lane), c);
    c = WMMA_BF16(a2, load_frag(Bbase, KCL, 64, lane), c);
    c = WMMA_BF16(a3, load_frag(Bbase, KCL, 96, lane), c);
    int n = (ct << 4) + nloc;
    #pragma unroll
    for (int i = 0; i < 8; ++i) {
      float cs = 1.0f / (1.0f + __expf(-c[i]));
      size_t idx = (size_t)(row0 + mbase + i) * HID + n;
      accp[i] += cs * pos[idx];
      accn[i] += cs * neg[idx];
    }
  }
  // reduce across the 16 lanes that share each row m (xor stays inside half)
  #pragma unroll
  for (int msk = 8; msk; msk >>= 1) {
    #pragma unroll
    for (int i = 0; i < 8; ++i) {
      accp[i] += __shfl_xor(accp[i], msk, 32);
      accn[i] += __shfl_xor(accn[i], msk, 32);
    }
  }
  if (nloc == 0) {
    float s3 = 0.f, s4 = 0.f;
    #pragma unroll
    for (int i = 0; i < 8; ++i) { s3 += softplusf(-accp[i]); s4 += softplusf(accn[i]); }
    atomicAdd(&acc[2], s3);
    atomicAdd(&acc[3], s4);
  }
}

// --------------------------- auxiliary kernels -----------------------------
__global__ void zero_f32_kernel(float* p, int n) {
  int i = blockIdx.x * blockDim.x + threadIdx.x;
  if (i < n) p[i] = 0.0f;
}

__global__ void degree_kernel(const int* __restrict__ src, const int* __restrict__ dst,
                              float* dout, float* din, int Et) {
  int e = blockIdx.x * blockDim.x + threadIdx.x;
  if (e >= Et) return;
  atomicAdd(&dout[src[e]], 1.0f);
  atomicAdd(&din[dst[e]],  1.0f);
}

__global__ void inv_sqrt_kernel(float* d, int n) {
  int i = blockIdx.x * blockDim.x + threadIdx.x;
  if (i < n) d[i] = rsqrtf(fmaxf(d[i], 1.0f));
}

// agg[dst[e]] += feat[perm?perm[src[e]]:src[e]] * inv_out[src[e]]; one block/edge
__global__ void scatter_kernel(const float* __restrict__ feat,
                               const int* __restrict__ perm,
                               const int* __restrict__ src,
                               const int* __restrict__ dst,
                               const float* __restrict__ inv_out,
                               float* __restrict__ agg, int Et)
{
  int e = blockIdx.x;
  if (e >= Et) return;
  int s = src[e];
  int srow = perm ? perm[s] : s;
  float w = inv_out[s];
  const float* fs = feat + (size_t)srow * HID;
  float* ag = agg + (size_t)dst[e] * HID;
  for (int h = threadIdx.x; h < HID; h += blockDim.x)
    atomicAdd(&ag[h], fs[h] * w);
}

__global__ void scale_cvt_kernel(const float* __restrict__ agg,
                                 const float* __restrict__ inv_in,
                                 __bf16* __restrict__ X, int total) {
  int i = blockIdx.x * blockDim.x + threadIdx.x;
  if (i >= total) return;
  X[i] = (__bf16)(agg[i] * inv_in[i >> 9]);  // H == 512
}

// Wt[c*rows + r] = (bf16) W[r*cols + c]
__global__ void transpose_bf16_kernel(const float* __restrict__ W,
                                      __bf16* __restrict__ Wt,
                                      int rows, int cols) {
  int i = blockIdx.x * blockDim.x + threadIdx.x;
  if (i >= rows * cols) return;
  int rI = i / cols, cI = i - rI * cols;
  Wt[(size_t)cI * rows + rI] = (__bf16)W[i];
}

__global__ void copy_f32_kernel(const float* a, float* b, int n) {
  int i = blockIdx.x * blockDim.x + threadIdx.x;
  if (i < n) b[i] = a[i];
}

__global__ void colmean_sigmoid_kernel(const float* __restrict__ pos,
                                       float* __restrict__ gs, int Nn) {
  __shared__ float sh[256];
  int h = blockIdx.x;
  float s = 0.f;
  for (int n = threadIdx.x; n < Nn; n += blockDim.x) s += pos[(size_t)n * HID + h];
  sh[threadIdx.x] = s; __syncthreads();
  for (int off = blockDim.x >> 1; off > 0; off >>= 1) {
    if ((int)threadIdx.x < off) sh[threadIdx.x] += sh[threadIdx.x + off];
    __syncthreads();
  }
  if (threadIdx.x == 0) gs[h] = 1.0f / (1.0f + __expf(-sh[0] / (float)Nn));
}

__global__ void matvec_kernel(const float* __restrict__ W,
                              const float* __restrict__ x,
                              float* __restrict__ y) {
  __shared__ float sh[256];
  int rowi = blockIdx.x;
  float s = 0.f;
  for (int j = threadIdx.x; j < HID; j += blockDim.x)
    s += W[(size_t)rowi * HID + j] * x[j];
  sh[threadIdx.x] = s; __syncthreads();
  for (int off = blockDim.x >> 1; off > 0; off >>= 1) {
    if ((int)threadIdx.x < off) sh[threadIdx.x] += sh[threadIdx.x + off];
    __syncthreads();
  }
  if (threadIdx.x == 0) y[rowi] = sh[0];
}

// one wave per node: dots against ws vector; softplus sums into acc[0..1]
__global__ void graph_loss_kernel(const float* __restrict__ pos,
                                  const float* __restrict__ neg,
                                  const float* __restrict__ wsv,
                                  float* __restrict__ acc, int Nn) {
  int node = blockIdx.x * (blockDim.x >> 5) + (threadIdx.x >> 5);
  if (node >= Nn) return;
  int lane = threadIdx.x & 31;
  float sp = 0.f, sn = 0.f;
  for (int h = lane; h < HID; h += 32) {
    float w = wsv[h];
    sp += pos[(size_t)node * HID + h] * w;
    sn += neg[(size_t)node * HID + h] * w;
  }
  for (int m = 16; m; m >>= 1) {
    sp += __shfl_xor(sp, m, 32);
    sn += __shfl_xor(sn, m, 32);
  }
  if (lane == 0) {
    atomicAdd(&acc[0], softplusf(-sp));
    atomicAdd(&acc[1], softplusf(sn));
  }
}

// data = pos / (||row||+1e-8), written bf16 row-major and transposed
__global__ void row_normalize_kernel(const float* __restrict__ pos,
                                     __bf16* __restrict__ data,
                                     __bf16* __restrict__ dataT, int Nn) {
  int node = blockIdx.x * (blockDim.x >> 5) + (threadIdx.x >> 5);
  if (node >= Nn) return;
  int lane = threadIdx.x & 31;
  const float* row = pos + (size_t)node * HID;
  float ss = 0.f;
  for (int h = lane; h < HID; h += 32) { float v = row[h]; ss += v * v; }
  for (int m = 16; m; m >>= 1) ss += __shfl_xor(ss, m, 32);
  float inv = 1.0f / (sqrtf(ss) + 1e-8f);
  for (int h = lane; h < HID; h += 32) {
    float v = row[h] * inv;
    data[(size_t)node * HID + h]  = (__bf16)v;
    dataT[(size_t)h * Nn + node]  = (__bf16)v;
  }
}

__global__ void mu_normalize_kernel(const float* __restrict__ mu,
                                    __bf16* __restrict__ munorm) {
  int c = blockIdx.x * (blockDim.x >> 5) + (threadIdx.x >> 5);
  if (c >= KCL) return;
  int lane = threadIdx.x & 31;
  const float* row = mu + (size_t)c * HID;
  float ss = 0.f;
  for (int h = lane; h < HID; h += 32) { float v = row[h]; ss += v * v; }
  for (int m = 16; m; m >>= 1) ss += __shfl_xor(ss, m, 32);
  float inv = rsqrtf(ss);
  for (int h = lane; h < HID; h += 32)
    munorm[(size_t)c * HID + h] = (__bf16)(row[h] * inv);
}

// one wave per node: r = softmax(BETA*dist); write r, rT, accumulate colsum
__global__ void softmax_kernel(const float* __restrict__ dist,
                               __bf16* __restrict__ r,
                               __bf16* __restrict__ rT,
                               float* __restrict__ colsum, int Nn) {
  int node = blockIdx.x * (blockDim.x >> 5) + (threadIdx.x >> 5);
  if (node >= Nn) return;
  int lane = threadIdx.x & 31;
  const float* drow = dist + (size_t)node * KCL;
  float x[4];
  float mx = -3.0e38f;
  #pragma unroll
  for (int j = 0; j < 4; ++j) { x[j] = BETA_T * drow[lane + 32 * j]; mx = fmaxf(mx, x[j]); }
  for (int m = 16; m; m >>= 1) mx = fmaxf(mx, __shfl_xor(mx, m, 32));
  float s = 0.f;
  #pragma unroll
  for (int j = 0; j < 4; ++j) { x[j] = __expf(x[j] - mx); s += x[j]; }
  for (int m = 16; m; m >>= 1) s += __shfl_xor(s, m, 32);
  float invs = 1.0f / s;
  #pragma unroll
  for (int j = 0; j < 4; ++j) {
    int idx = lane + 32 * j;
    float rv = x[j] * invs;
    r[(size_t)node * KCL + idx]  = (__bf16)rv;
    rT[(size_t)idx * Nn + node]  = (__bf16)rv;
    atomicAdd(&colsum[idx], rv);
  }
}

__global__ void recip_kernel(const float* a, float* b, int n) {
  int i = blockIdx.x * blockDim.x + threadIdx.x;
  if (i < n) b[i] = 1.0f / a[i];
}

__global__ void finalize_kernel(const float* acc, float* out, int Nn) {
  float invN = 1.0f / (float)Nn;
  out[0] = 0.5f * (acc[0] + acc[1]) * invN + 0.5f * (acc[2] + acc[3]) * invN;
}

// ---------------------------------------------------------------------------
extern "C" void kernel_launch(void* const* d_in, const int* in_sizes, int n_in,
                              void* d_out, int out_size, void* d_ws, size_t ws_size,
                              hipStream_t stream) {
  const float* features     = (const float*)d_in[0];
  const float* W1           = (const float*)d_in[1];
  const float* b1           = (const float*)d_in[2];
  const float* W2           = (const float*)d_in[3];
  const float* b2           = (const float*)d_in[4];
  const float* disc_W       = (const float*)d_in[5];
  const float* cluster_init = (const float*)d_in[6];
  const int*   edge_index   = (const int*)d_in[7];
  const int*   perm         = (const int*)d_in[8];

  const int Et = in_sizes[7] / 2;     // 340000
  const int Nn = in_sizes[8];         // 20000 (multiple of 32)
  const int* src = edge_index;
  const int* dst = edge_index + Et;

  // ---- workspace carve ----
  char* p = (char*)d_ws;
  auto take = [&](size_t bytes) -> void* {
    void* r = (void*)p;
    p += (bytes + 255) & ~(size_t)255;
    return r;
  };
  float*  pos   = (float*)take(sizeof(float) * Nn * HID);
  float*  neg   = (float*)take(sizeof(float) * Nn * HID);
  float*  agg   = (float*)take(sizeof(float) * Nn * HID);
  float*  hbuf  = (float*)take(sizeof(float) * Nn * HID);
  __bf16* Xbf   = (__bf16*)take(sizeof(__bf16) * Nn * HID);
  __bf16* W1t   = (__bf16*)take(sizeof(__bf16) * DIMF * HID);
  __bf16* W2t   = (__bf16*)take(sizeof(__bf16) * HID * HID);
  float*  mu    = (float*)take(sizeof(float) * KCL * HID);
  __bf16* munorm= (__bf16*)take(sizeof(__bf16) * KCL * HID);
  __bf16* muT   = (__bf16*)take(sizeof(__bf16) * HID * KCL);
  float*  invO  = (float*)take(sizeof(float) * Nn);
  float*  invI  = (float*)take(sizeof(float) * Nn);
  float*  gs    = (float*)take(sizeof(float) * HID);
  float*  wsv   = (float*)take(sizeof(float) * HID);
  float*  colsum= (float*)take(sizeof(float) * KCL);
  float*  rowdiv= (float*)take(sizeof(float) * KCL);
  float*  acc   = (float*)take(sizeof(float) * 8);
  // aliases (valid after the GCN passes finish):
  float*  dist  = agg;                                             // Nn*KCL f32
  __bf16* rbf   = (__bf16*)((char*)agg + ((sizeof(float) * Nn * KCL + 255) & ~(size_t)255));
  __bf16* rT    = (__bf16*)((char*)rbf + ((sizeof(__bf16) * Nn * KCL + 255) & ~(size_t)255));
  __bf16* data  = Xbf;
  __bf16* dataT = (__bf16*)hbuf;

  auto zero = [&](float* q, int n) {
    zero_f32_kernel<<<(n + 255) / 256, 256, 0, stream>>>(q, n);
  };
  auto gemm = [&](const __bf16* A, int lda, const __bf16* Bt, int ldb,
                  float* C, int ldc, int M, int Ncol, int Kd,
                  const float* bias, const float* rdv, int mode) {
    int tiles = (M >> 5) * (Ncol >> 5);  // 32x32 supertiles, 1 wave each
    gemm_bf16_kernel<<<(tiles + 7) / 8, 256, 0, stream>>>(
        A, lda, Bt, ldb, C, ldc, M, Ncol, Kd, bias, rdv, mode);
  };

  // ---- degree normalization ----
  zero(acc, 8);
  zero(invO, Nn);
  zero(invI, Nn);
  degree_kernel<<<(Et + 255) / 256, 256, 0, stream>>>(src, dst, invO, invI, Et);
  inv_sqrt_kernel<<<(Nn + 255) / 256, 256, 0, stream>>>(invO, Nn);
  inv_sqrt_kernel<<<(Nn + 255) / 256, 256, 0, stream>>>(invI, Nn);

  // ---- weights -> bf16 transposed (Bt form) ----
  transpose_bf16_kernel<<<(DIMF * HID + 255) / 256, 256, 0, stream>>>(W1, W1t, DIMF, HID);
  transpose_bf16_kernel<<<(HID * HID + 255) / 256, 256, 0, stream>>>(W2, W2t, HID, HID);

  // ---- two-layer GCN encoder, pos (identity) and neg (permuted rows) ----
  for (int pass = 0; pass < 2; ++pass) {
    const int* pp = (pass == 0) ? (const int*)nullptr : perm;
    float* outb = (pass == 0) ? pos : neg;
    zero(agg, Nn * HID);
    scatter_kernel<<<Et, 256, 0, stream>>>(features, pp, src, dst, invO, agg, Et);
    scale_cvt_kernel<<<(Nn * HID + 255) / 256, 256, 0, stream>>>(agg, invI, Xbf, Nn * HID);
    gemm(Xbf, HID, W1t, DIMF, hbuf, HID, Nn, HID, DIMF, b1, nullptr, 1); // +b1, relu
    zero(agg, Nn * HID);
    scatter_kernel<<<Et, 256, 0, stream>>>(hbuf, nullptr, src, dst, invO, agg, Et);
    scale_cvt_kernel<<<(Nn * HID + 255) / 256, 256, 0, stream>>>(agg, invI, Xbf, Nn * HID);
    gemm(Xbf, HID, W2t, HID, outb, HID, Nn, HID, HID, b2, nullptr, 2);   // +b2
  }

  // ---- graph summary + bilinear graph discriminator loss ----
  colmean_sigmoid_kernel<<<HID, 256, 0, stream>>>(pos, gs, Nn);
  matvec_kernel<<<HID, 256, 0, stream>>>(disc_W, gs, wsv);
  graph_loss_kernel<<<(Nn + 7) / 8, 256, 0, stream>>>(pos, neg, wsv, acc, Nn);

  // ---- soft k-means (11 total iterations: 10 + 1) ----
  row_normalize_kernel<<<(Nn + 7) / 8, 256, 0, stream>>>(pos, data, dataT, Nn);
  copy_f32_kernel<<<(KCL * HID + 255) / 256, 256, 0, stream>>>(cluster_init, mu, KCL * HID);
  for (int it = 0; it < CITER; ++it) {
    mu_normalize_kernel<<<(KCL + 7) / 8, 256, 0, stream>>>(mu, munorm);
    gemm(data, HID, munorm, HID, dist, KCL, Nn, KCL, HID, nullptr, nullptr, 0);
    zero(colsum, KCL);
    softmax_kernel<<<(Nn + 7) / 8, 256, 0, stream>>>(dist, rbf, rT, colsum, Nn);
    recip_kernel<<<1, KCL, 0, stream>>>(colsum, rowdiv, KCL);
    gemm(rT, Nn, dataT, Nn, mu, HID, KCL, HID, Nn, nullptr, rowdiv, 3);  // /colsum
  }

  // ---- fused cluster discriminator loss (r @ mu -> sigmoid -> dots) ----
  transpose_bf16_kernel<<<(KCL * HID + 255) / 256, 256, 0, stream>>>(mu, muT, KCL, HID);
  cluster_loss_kernel<<<((Nn >> 4) + 7) / 8, 256, 0, stream>>>(rbf, muT, pos, neg, acc, Nn);

  finalize_kernel<<<1, 1, 0, stream>>>(acc, (float*)d_out, Nn);
}
